// S2Net_69965017252363
// MI455X (gfx1250) — compile-verified, matrix operand order checked by reading, not run on hardware
//
#include <hip/hip_runtime.h>
#include <hip/hip_bf16.h>

typedef __attribute__((ext_vector_type(16))) _Float16 v16h;
typedef __attribute__((ext_vector_type(8)))  _Float16 v8h;
typedef __attribute__((ext_vector_type(8)))  float    v8f;

// Problem constants (match reference)
#define BB 32
#define TT 512
#define NN 512
#define DD 4
#define HH 2048
#define CC 128
#define DT_C    0.1f
#define DECAY_C 0.5f
#define VTH_C   1.0f
#define ALPHA_C 4.0f
#define NEG_C   0.1f
#define EPS_C   1e-5f
#define KOVN_C  (1.0f / 512.0f)   // K_COUP / N

// ------------------------------------------------------------------
// K-index permutation for the 16-bit WMMA A/B operand layout:
//   element e, lane-half h  ->  K = e + 8h + (e>=8 ? 8 : 0)
// Inverse (raw k within a 32-chunk -> slot h*16+e): swap the middle octets.
// ------------------------------------------------------------------
__device__ __forceinline__ int permpos(int r) {
    int oct = r >> 3;
    return r + (oct == 1 ? 8 : (oct == 2 ? -8 : 0));
}

// ------------------------------------------------------------------
// Weight pre-pack kernels: fp32 row-major [K][J] -> f16 B-operand tiles.
// dst index = ((jt*KC + kc)*32 + lane)*16 + e ; lane = h*16 + n
// value = W[kc*32 + e + 8h + (e>=8?8:0)][jt*16 + n]
// ------------------------------------------------------------------
__global__ void pack_b_kernel(const float* __restrict__ src,
                              _Float16* __restrict__ dst, int K, int J) {
    const int KC = K >> 5;
    const size_t total = (size_t)K * (size_t)J;
    for (size_t idx = (size_t)blockIdx.x * blockDim.x + threadIdx.x;
         idx < total; idx += (size_t)gridDim.x * blockDim.x) {
        int e    = (int)(idx & 15);
        int lane = (int)((idx >> 4) & 31);
        size_t chunk = idx >> 9;
        int kc = (int)(chunk % KC);
        int jt = (int)(chunk / KC);
        int h = lane >> 4, n = lane & 15;
        int k = kc * 32 + e + 8 * h + (e >= 8 ? 8 : 0);
        dst[idx] = (_Float16)src[(size_t)k * J + jt * 16 + n];
    }
}

// sc -> A-operand tiles: lane = h*16 + m (row), value = sc[it*16+m][k]
__global__ void pack_a_kernel(const float* __restrict__ src,
                              _Float16* __restrict__ dst, int Nrows, int K) {
    const int KC = K >> 5;
    const size_t total = (size_t)Nrows * (size_t)K;
    for (size_t idx = (size_t)blockIdx.x * blockDim.x + threadIdx.x;
         idx < total; idx += (size_t)gridDim.x * blockDim.x) {
        int e    = (int)(idx & 15);
        int lane = (int)((idx >> 4) & 31);
        size_t chunk = idx >> 9;
        int kc = (int)(chunk % KC);
        int it = (int)(chunk / KC);
        int h = lane >> 4, m = lane & 15;
        int k = kc * 32 + e + 8 * h + (e >= 8 ? 8 : 0);
        dst[idx] = (_Float16)src[(size_t)(it * 16 + m) * K + k];
    }
}

// ------------------------------------------------------------------
// Broadcast-A GEMV tile: y[jt*16 .. +15] = x . W  via WMMA f16->f32.
// xp = LDS activation, permuted per 32-K chunk (slot = h*16+e).
// Wp = packed B-operand weights. Result valid in lanes 0..15, acc[0].
// ------------------------------------------------------------------
__device__ __forceinline__ v8f gemv_tile(const _Float16* __restrict__ Wp,
                                         const _Float16* __restrict__ xp,
                                         int KC, int jt, int lane) {
    v8f acc;
#pragma unroll
    for (int i = 0; i < 8; ++i) acc[i] = 0.0f;
    const int hoff = lane & 16;
    const _Float16* wb = Wp + (size_t)jt * KC * 512 + (size_t)lane * 16;
    for (int kc = 0; kc < KC; ++kc) {
        v16h a = *(const v16h*)(xp + kc * 32 + hoff);
        v16h b = *(const v16h*)(wb + (size_t)kc * 512);
        acc = __builtin_amdgcn_wmma_f32_16x16x32_f16(
            false, a, false, b, (short)0, acc, false, false);
    }
    return acc;
}

// Async global->LDS copy of 16 bytes per active lane (ASYNCcnt tracked).
__device__ __forceinline__ void async_copy_b128(unsigned lds_off,
                                                const float* gaddr) {
    asm volatile("global_load_async_to_lds_b128 %0, %1, off"
                 :: "v"(lds_off), "v"(gaddr) : "memory");
}
__device__ __forceinline__ void wait_asynccnt0() {
    asm volatile("s_wait_asynccnt 0x0" ::: "memory");
}

// ------------------------------------------------------------------
// Persistent scan kernel: one workgroup per batch element, 8 waves.
// ------------------------------------------------------------------
__global__ void __launch_bounds__(256, 1)
s2net_scan_kernel(const float* __restrict__ x,
                  const float* __restrict__ b_proj,
                  const float* __restrict__ ln_g,
                  const float* __restrict__ ln_b,
                  const float* __restrict__ b_enc,
                  const float* __restrict__ omega,
                  const float* __restrict__ b_mask,
                  const float* __restrict__ b_in,
                  const float* __restrict__ b_out,
                  const _Float16* __restrict__ Wp,   // proj   K=512  J=2048
                  const _Float16* __restrict__ We,   // enc    K=2048 J=512
                  const _Float16* __restrict__ Wm,   // mask   K=512  J=512
                  const _Float16* __restrict__ Wi,   // in     K=2048 J=2048
                  const _Float16* __restrict__ Wo,   // out    K=2048 J=128
                  const _Float16* __restrict__ scA,  // sc packed as A tiles
                  float* __restrict__ snn,           // [B][T][C] workspace
                  float* __restrict__ spikes)        // [B][T][H] output
{
    __shared__ float theta[NN * DD];                  // 8 KB  persistent
    __shared__ float p2m[NN * DD];                    // 8 KB  theta delay slot
    __shared__ float memv[HH];                        // 8 KB  membrane
    __shared__ float coupz[HH];                       // 8 KB  z / coup / sin(theta_new)
    __shared__ __align__(16) float xstage[2][NN];     // 4 KB  async x double buffer
    __shared__ __align__(32) _Float16 stctT[8 * NN];  // 8 KB  row n: n<4 sin_d / n>=4 cos_d
    __shared__ __align__(32) _Float16 xperm[NN];      // 1 KB
    __shared__ __align__(32) _Float16 hbuf[HH];       // 4 KB  zh, then feat
    __shared__ __align__(32) _Float16 sbuf[HH];       // 4 KB  spike f16
    __shared__ __align__(32) _Float16 gperm[NN];      // 1 KB
    __shared__ float gammal[NN];                      // 2 KB
    __shared__ float gl[NN];                          // 2 KB  g (also LN reduce)
    __shared__ float maskl[NN];                       // 2 KB  mask (also LN reduce)

    const int tid  = threadIdx.x;
    const int lane = tid & 31;
    const int wave = tid >> 5;
    const int b    = blockIdx.x;
    const int h16  = lane >> 4;     // lane half
    const int ncol = lane & 15;     // column within tile

    // Thread-owned flat indices f = tid*8 .. tid*8+7  (covers 2048 = N*D = H)
    float om[8];
#pragma unroll
    for (int q = 0; q < 8; ++q) {
        int f = tid * 8 + q;
        om[q] = omega[f];
        theta[f] = 0.0f;
        p2m[f]   = 0.0f;
        memv[f]  = 0.0f;
    }
    // Kick off async copy of x_0 into staging buffer 0 (waves 0-3, 16B/lane).
    if (tid < 128) {
        async_copy_b128((unsigned)(uintptr_t)&xstage[0][tid * 4],
                        x + (size_t)b * TT * NN + tid * 4);
    }
    __syncthreads();

    for (int t = 0; t < TT; ++t) {
        const int cur = t & 1, nxt = cur ^ 1;
        // Publish xstage[cur]: issuing waves drain ASYNCcnt, then barrier.
        if (tid < 128) wait_asynccnt0();
        __syncthreads();
        // Prefetch x_{t+1} into the other buffer (its last reader was step t-1).
        if (tid < 128 && t + 1 < TT) {
            async_copy_b128((unsigned)(uintptr_t)&xstage[nxt][tid * 4],
                            x + ((size_t)b * TT + (t + 1)) * NN + tid * 4);
        }

        // ---- Stage 1: x_t -> xperm (f16, K-permuted); stctT from theta_old
        {
#pragma unroll
            for (int q = 0; q < 2; ++q) {
                int j = tid * 2 + q;
                xperm[(j >> 5) * 32 + permpos(j & 31)] = (_Float16)xstage[cur][j];
            }
#pragma unroll
            for (int q = 0; q < 8; ++q) {
                int f = tid * 8 + q;
                int j = f >> 2, d = f & 3;
                float th = theta[f];
                stctT[d * NN + j]       = (_Float16)__sinf(th);
                stctT[(4 + d) * NN + j] = (_Float16)__cosf(th);
            }
        }
        __syncthreads();

        // ---- Stage 2: z = x @ W_proj + b_proj   (128 j-tiles, KC=16)
        for (int jt = wave; jt < 128; jt += 8) {
            v8f acc = gemv_tile(Wp, xperm, 16, jt, lane);
            if (lane < 16) {
                int j = jt * 16 + ncol;
                coupz[j] = acc[0] + b_proj[j];    // coupz doubles as z buffer
            }
        }
        __syncthreads();

        // ---- Stage 3: LayerNorm + LeakyReLU -> hbuf (f16, permuted)
        {
            float s1 = 0.0f, s2 = 0.0f;
#pragma unroll
            for (int q = 0; q < 8; ++q) {
                float v = coupz[tid * 8 + q];
                s1 += v; s2 += v * v;
            }
            gl[tid] = s1; maskl[tid] = s2;
            __syncthreads();
            for (int s = 128; s > 0; s >>= 1) {
                if (tid < s) { gl[tid] += gl[tid + s]; maskl[tid] += maskl[tid + s]; }
                __syncthreads();
            }
            float mu  = gl[0] * (1.0f / HH);
            float var = maskl[0] * (1.0f / HH) - mu * mu;
            float rs  = rsqrtf(var + EPS_C);
#pragma unroll
            for (int q = 0; q < 8; ++q) {
                int j = tid * 8 + q;
                float v = (coupz[j] - mu) * rs * ln_g[j] + ln_b[j];
                v = (v > 0.0f) ? v : NEG_C * v;
                hbuf[(j >> 5) * 32 + permpos(j & 31)] = (_Float16)v;
            }
        }
        __syncthreads();

        // ---- Stage 4: gamma = z_ln @ W_enc + b_enc  (32 j-tiles, KC=64)
        for (int jt = wave; jt < 32; jt += 8) {
            v8f acc = gemv_tile(We, hbuf, 64, jt, lane);
            if (lane < 16) {
                int j = jt * 16 + ncol;
                gammal[j] = acc[0] + b_enc[j];
            }
        }
        __syncthreads();

        // ---- Stage 5: coupling GEMM  sc @ [sin|cos]  (32 M-tiles, KC=16)
        // B operand: column n<4 = sin_d, n in 4..7 = cos_d, n>=8 = 0.
        // stctT transposed rows -> two contiguous ds_load_b128 per chunk.
        for (int it = wave; it < 32; it += 8) {
            v8f acc;
#pragma unroll
            for (int i = 0; i < 8; ++i) acc[i] = 0.0f;
            v16h zer;
#pragma unroll
            for (int e = 0; e < 16; ++e) zer[e] = (_Float16)0.0f;
            const int nn7 = ncol & 7;
            const _Float16* ab = scA + (size_t)it * 16 * 512 + (size_t)lane * 16;
            for (int kc = 0; kc < 16; ++kc) {
                v16h a = *(const v16h*)(ab + (size_t)kc * 512);
                const _Float16* srow = stctT + nn7 * NN + kc * 32 + 8 * h16;
                v8h blo = *(const v8h*)(srow);        // K = base .. base+7
                v8h bhi = *(const v8h*)(srow + 16);   // K = base+16 .. base+23
                v16h bb = __builtin_shufflevector(blo, bhi,
                    0, 1, 2, 3, 4, 5, 6, 7, 8, 9, 10, 11, 12, 13, 14, 15);
                bb = (ncol < 8) ? bb : zer;
                acc = __builtin_amdgcn_wmma_f32_16x16x32_f16(
                    false, a, false, bb, (short)0, acc, false, false);
            }
            // coup[i][d] = cos*C_sin - sin*C_cos ; column d holds C_sin, d+4 C_cos
#pragma unroll
            for (int r = 0; r < 8; ++r) {
                float cr = __shfl_xor(acc[r], 4);     // partner column's value
                if (ncol < 4) {
                    int i = it * 16 + r + 8 * h16;
                    float st_ = (float)stctT[ncol * NN + i];
                    float ct_ = (float)stctT[(4 + ncol) * NN + i];
                    coupz[i * 4 + ncol] = ct_ * acc[r] - st_ * cr;
                }
            }
        }
        __syncthreads();

        // ---- Stage 6: theta update, delay line, g, sin(theta_new)
        {
            float dsum0 = 0.0f, dsum1 = 0.0f;
#pragma unroll
            for (int q = 0; q < 8; ++q) {
                int f = tid * 8 + q;
                int i = f >> 2;
                float thold = theta[f];
                float thn = thold + DT_C * (om[q] + gammal[i] + KOVN_C * coupz[f]);
                float delayed = (t == 0) ? thn : p2m[f];
                p2m[f] = (t == 0) ? thn : thold;
                theta[f] = thn;
                coupz[f] = __sinf(thn);               // reuse as sin(theta_new)
                if (q < 4) dsum0 += delayed; else dsum1 += delayed;
            }
            int i0 = tid * 2;
            float g0 = 0.5f * (1.0f + __sinf(dsum0 * 0.25f));
            float g1 = 0.5f * (1.0f + __sinf(dsum1 * 0.25f));
            gl[i0] = g0; gl[i0 + 1] = g1;
            gperm[(i0 >> 5) * 32 + permpos(i0 & 31)] = (_Float16)g0;
            int i1 = i0 + 1;
            gperm[(i1 >> 5) * 32 + permpos(i1 & 31)] = (_Float16)g1;
        }
        __syncthreads();

        // ---- Stage 7: mask = sigmoid(g @ W_mask + b_mask)  (32 j-tiles, KC=16)
        for (int jt = wave; jt < 32; jt += 8) {
            v8f acc = gemv_tile(Wm, gperm, 16, jt, lane);
            if (lane < 16) {
                int j = jt * 16 + ncol;
                maskl[j] = 1.0f / (1.0f + __expf(-(acc[0] + b_mask[j])));
            }
        }
        __syncthreads();

        // ---- Stage 8: feat = sin(theta_new) * g * mask -> hbuf (f16 permuted)
#pragma unroll
        for (int q = 0; q < 8; ++q) {
            int f = tid * 8 + q;
            int i = f >> 2;
            float v = coupz[f] * gl[i] * maskl[i];
            hbuf[(f >> 5) * 32 + permpos(f & 31)] = (_Float16)v;
        }
        __syncthreads();

        // ---- Stage 9: cur = feat @ W_in + b_in ; LIF update ; spikes out (NT)
        for (int jt = wave; jt < 128; jt += 8) {
            v8f acc = gemv_tile(Wi, hbuf, 64, jt, lane);
            if (lane < 16) {
                int j = jt * 16 + ncol;
                float cu = acc[0] + b_in[j];
                float m2 = DECAY_C * memv[j] + cu;
                float sp = 1.0f / (1.0f + __expf(-ALPHA_C * (m2 - VTH_C)));
                memv[j] = m2 - sp * VTH_C;
                // Spikes are never re-read on device: non-temporal store.
                __builtin_nontemporal_store(sp,
                    &spikes[((size_t)b * TT + t) * HH + j]);
                sbuf[(j >> 5) * 32 + permpos(j & 31)] = (_Float16)sp;
            }
        }
        __syncthreads();

        // ---- Stage 10: out_t = spike @ W_out + b_out (8 j-tiles, KC=64)
        {
            v8f acc = gemv_tile(Wo, sbuf, 64, wave, lane);
            if (lane < 16) {
                int c0 = wave * 16 + ncol;
                snn[((size_t)b * TT + t) * CC + c0] = acc[0] + b_out[c0];
            }
        }
        __syncthreads();   // protects xperm/stctT/buffers for next step
    }
}

// ------------------------------------------------------------------
// Conv1d (pad=2, kernel=5) + log_softmax over C. One block per (t, b).
// snn stored [B][T][C]; out is [B][C][T].
// ------------------------------------------------------------------
__global__ void conv_lsm_kernel(const float* __restrict__ snn,
                                const float* __restrict__ Wc,
                                const float* __restrict__ bc,
                                float* __restrict__ out) {
    const int t = blockIdx.x, b = blockIdx.y, c = threadIdx.x;
    __shared__ float rows[5][CC];
    __shared__ float red[CC];
#pragma unroll
    for (int k = 0; k < 5; ++k) {
        int tp = t + k - 2;
        rows[k][c] = (tp >= 0 && tp < TT) ? snn[((size_t)b * TT + tp) * CC + c] : 0.0f;
    }
    __syncthreads();
    float acc = bc[c];
#pragma unroll
    for (int k = 0; k < 5; ++k) {
        const float* w = Wc + (size_t)c * CC * 5 + k;
        for (int cp = 0; cp < CC; ++cp)
            acc += rows[k][cp] * w[(size_t)cp * 5];
    }
    red[c] = acc; __syncthreads();
    for (int s = 64; s > 0; s >>= 1) {
        if (c < s) red[c] = fmaxf(red[c], red[c + s]);
        __syncthreads();
    }
    float mx = red[0]; __syncthreads();
    red[c] = __expf(acc - mx); __syncthreads();
    for (int s = 64; s > 0; s >>= 1) {
        if (c < s) red[c] += red[c + s];
        __syncthreads();
    }
    float lse = mx + __logf(red[0]);
    out[((size_t)b * CC + c) * TT + t] = acc - lse;
}

// ------------------------------------------------------------------
extern "C" void kernel_launch(void* const* d_in, const int* in_sizes, int n_in,
                              void* d_out, int out_size, void* d_ws, size_t ws_size,
                              hipStream_t stream) {
    const float* x      = (const float*)d_in[0];
    const float* sc     = (const float*)d_in[1];
    const float* W_proj = (const float*)d_in[2];
    const float* b_proj = (const float*)d_in[3];
    const float* ln_g   = (const float*)d_in[4];
    const float* ln_b   = (const float*)d_in[5];
    const float* W_enc  = (const float*)d_in[6];
    const float* b_enc  = (const float*)d_in[7];
    const float* omega  = (const float*)d_in[8];
    const float* W_mask = (const float*)d_in[9];
    const float* b_mask = (const float*)d_in[10];
    const float* W_in   = (const float*)d_in[11];
    const float* b_in   = (const float*)d_in[12];
    const float* W_out  = (const float*)d_in[13];
    const float* b_out  = (const float*)d_in[14];
    const float* W_conv = (const float*)d_in[15];
    const float* b_conv = (const float*)d_in[16];

    // Workspace layout (bytes)
    char* ws = (char*)d_ws;
    size_t o = 0;
    float*    snn  = (float*)(ws + o);    o += (size_t)BB * TT * CC * 4;      // 8 MB
    _Float16* WpP  = (_Float16*)(ws + o); o += (size_t)512  * 2048 * 2;       // 2 MB
    _Float16* WeP  = (_Float16*)(ws + o); o += (size_t)2048 * 512  * 2;       // 2 MB
    _Float16* WmP  = (_Float16*)(ws + o); o += (size_t)512  * 512  * 2;       // 0.5 MB
    _Float16* WiP  = (_Float16*)(ws + o); o += (size_t)2048 * 2048 * 2;       // 8 MB
    _Float16* WoP  = (_Float16*)(ws + o); o += (size_t)2048 * 128  * 2;       // 0.5 MB
    _Float16* scAP = (_Float16*)(ws + o); o += (size_t)512  * 512  * 2;       // 0.5 MB

    // Pack weights into WMMA operand layouts (f32 -> f16)
    pack_b_kernel<<<2048, 256, 0, stream>>>(W_proj, WpP, 512,  2048);
    pack_b_kernel<<<2048, 256, 0, stream>>>(W_enc,  WeP, 2048, 512);
    pack_b_kernel<<<1024, 256, 0, stream>>>(W_mask, WmP, 512,  512);
    pack_b_kernel<<<4096, 256, 0, stream>>>(W_in,   WiP, 2048, 2048);
    pack_b_kernel<<<1024, 256, 0, stream>>>(W_out,  WoP, 2048, 128);
    pack_a_kernel<<<1024, 256, 0, stream>>>(sc,     scAP, 512, 512);

    float* out_lsm    = (float*)d_out;                          // [B][C][T]
    float* out_spikes = out_lsm + (size_t)BB * CC * TT;         // [B][T][H]

    // Persistent scan: one workgroup per batch element
    s2net_scan_kernel<<<BB, 256, 0, stream>>>(
        x, b_proj, ln_g, ln_b, b_enc, omega, b_mask, b_in, b_out,
        WpP, WeP, WmP, WiP, WoP, scAP, snn, out_spikes);

    // Temporal smoothing + log_softmax
    conv_lsm_kernel<<<dim3(TT, BB), CC, 0, stream>>>(snn, W_conv, b_conv, out_lsm);
}
